// PointNet2Encoder_26422638805440
// MI455X (gfx1250) — compile-verified, matrix operand order checked by reading, not run on hardware
//
#include <hip/hip_runtime.h>
#include <hip/hip_bf16.h>

typedef __attribute__((ext_vector_type(16))) _Float16 v16h;
typedef __attribute__((ext_vector_type(8)))  _Float16 v8h;
typedef __attribute__((ext_vector_type(8)))  float    v8f;
typedef __attribute__((ext_vector_type(4)))  unsigned int v4u;
typedef __attribute__((ext_vector_type(8)))  int      v8i;
typedef __attribute__((ext_vector_type(4)))  int      v4i;

#define BATCH 16
#define NPTS  8192
#define FPS_T 256

// -------------------------------------------------------------------------
// Farthest point sampling: one block per batch, dist[] in LDS. Emitted index
// is the incoming 'far' (fidx[0]==0); argmax tie-break = lowest index.
// -------------------------------------------------------------------------
__global__ void fps_kernel(const float* __restrict__ pos, int pstride, int N,
                           int npoint, int* __restrict__ fidx) {
  __shared__ float dist[NPTS];
  __shared__ float rmax[FPS_T];
  __shared__ int   ridx[FPS_T];
  const int b = blockIdx.x;
  const float* P = pos + (long)b * N * pstride;
  for (int i = threadIdx.x; i < N; i += FPS_T) dist[i] = 1e10f;
  __syncthreads();
  int far = 0;
  for (int step = 0; step < npoint; ++step) {
    if (threadIdx.x == 0) fidx[b * npoint + step] = far;
    const float cx = P[(long)far * pstride + 0];
    const float cy = P[(long)far * pstride + 1];
    const float cz = P[(long)far * pstride + 2];
    float bm = -1.0f; int bi = 0;
    for (int i = threadIdx.x; i < N; i += FPS_T) {
      float dx = P[(long)i * pstride + 0] - cx;
      float dy = P[(long)i * pstride + 1] - cy;
      float dz = P[(long)i * pstride + 2] - cz;
      float d = fminf(dist[i], dx * dx + dy * dy + dz * dz);
      dist[i] = d;
      if (d > bm) { bm = d; bi = i; }
    }
    rmax[threadIdx.x] = bm; ridx[threadIdx.x] = bi;
    __syncthreads();
    for (int off = FPS_T >> 1; off > 0; off >>= 1) {
      if (threadIdx.x < off) {
        float v2 = rmax[threadIdx.x + off]; int i2 = ridx[threadIdx.x + off];
        if (v2 > rmax[threadIdx.x] ||
            (v2 == rmax[threadIdx.x] && i2 < ridx[threadIdx.x])) {
          rmax[threadIdx.x] = v2; ridx[threadIdx.x] = i2;
        }
      }
      __syncthreads();
    }
    far = ridx[0];
    __syncthreads();
  }
}

__global__ void gather_centroids(const float* __restrict__ pos, int pstride, int N,
                                 const int* __restrict__ fidx,
                                 float* __restrict__ nx, int BS, int S) {
  int i = blockIdx.x * blockDim.x + threadIdx.x;
  if (i >= BS) return;
  int b = i / S;
  int j = fidx[i];
  const float* p = pos + ((long)b * N + j) * pstride;
  nx[i * 3 + 0] = p[0]; nx[i * 3 + 1] = p[1]; nx[i * 3 + 2] = p[2];
}

// -------------------------------------------------------------------------
// Ball query: one wave32 per query; ballot + prefix-popc compaction keeps
// indices ascending; tail padded with first hit.
// -------------------------------------------------------------------------
__global__ void ball_query_kernel(const float* __restrict__ pos, int pstride, int N,
                                  const float* __restrict__ new_xyz,
                                  int* __restrict__ gidx,
                                  int BS, int S, int nsample, float r2) {
  const int wid  = (blockIdx.x * blockDim.x + threadIdx.x) >> 5;
  const int lane = threadIdx.x & 31;
  if (wid >= BS) return;
  const int b = wid / S;
  const float qx = new_xyz[wid * 3 + 0];
  const float qy = new_xyz[wid * 3 + 1];
  const float qz = new_xyz[wid * 3 + 2];
  const float* P = pos + (long)b * N * pstride;
  int* out = gidx + (long)wid * nsample;
  const unsigned lt = (1u << lane) - 1u;
  int cnt = 0, first = -1;
  for (int n0 = 0; n0 < N && cnt < nsample; n0 += 32) {
    const int n = n0 + lane;
    bool in = false;
    if (n < N) {
      float dx = P[(long)n * pstride + 0] - qx;
      float dy = P[(long)n * pstride + 1] - qy;
      float dz = P[(long)n * pstride + 2] - qz;
      in = (dx * dx + dy * dy + dz * dz) <= r2;
    }
    unsigned mask = (unsigned)__ballot(in);
    if (in) {
      int p = cnt + __popc(mask & lt);
      if (p < nsample) out[p] = n;
    }
    if (first < 0 && mask) first = n0 + __builtin_ctz(mask);
    cnt += __popc(mask);
  }
  if (first < 0) first = 0;
  for (int p = cnt + lane; p < nsample; p += 32) out[p] = first;
}

// -------------------------------------------------------------------------
// Grouping -> f16 GEMM operand, row stride Kpad (mult of 32), zero padded.
// -------------------------------------------------------------------------
__global__ void gather_group_f16(const float* __restrict__ pos, int pstride,
                                 const float* __restrict__ feat, int fstride, int Cf,
                                 const float* __restrict__ new_xyz,
                                 const int* __restrict__ gidx,
                                 _Float16* __restrict__ X, int N, int S, int K,
                                 int Kpad, long total) {
  long i = (long)blockIdx.x * blockDim.x + threadIdx.x;   // over B*S*K
  if (i >= total) return;
  int k = (int)(i % K);
  long sk = i / K;                                         // b*S + s
  int b = (int)(sk / S);
  int j = gidx ? gidx[i] : k;
  const float* p = pos + ((long)b * N + j) * pstride;
  float cx = 0.f, cy = 0.f, cz = 0.f;
  if (new_xyz) { const float* q = new_xyz + sk * 3; cx = q[0]; cy = q[1]; cz = q[2]; }
  _Float16* xr = X + i * Kpad;
  xr[0] = (_Float16)(p[0] - cx);
  xr[1] = (_Float16)(p[1] - cy);
  xr[2] = (_Float16)(p[2] - cz);
  const float* f = feat + ((long)b * N + j) * fstride;
  for (int c = 0; c < Cf; ++c) xr[3 + c] = (_Float16)f[c];
  for (int c = 3 + Cf; c < Kpad; ++c) xr[c] = (_Float16)0.f;
}

__global__ void zero_f32(float* __restrict__ p, int n) {
  int i = blockIdx.x * blockDim.x + threadIdx.x;
  if (i < n) p[i] = 0.f;
}

// -------------------------------------------------------------------------
// Pack W (Kreal x C, f32) into per-lane WMMA B layout, f16, K padded to Kpad.
// Flat layout [kt][nt][lane][16 halves]: each lane's B fragment is one
// contiguous 32-byte run; a (kt, 4-tile group) row is 2048 contiguous halves.
// -------------------------------------------------------------------------
__global__ void pack_w(const float* __restrict__ W, _Float16* __restrict__ Wp,
                       int Kreal, int Kpad, int C) {
  long i = (long)blockIdx.x * blockDim.x + threadIdx.x;   // Kpad*C total
  if (i >= (long)Kpad * C) return;
  int h = (int)(i & 15); long t = i >> 4;
  int lane = (int)(t & 31); t >>= 5;
  int tilesN = C >> 4;
  int nt = (int)(t % tilesN);
  int kt = (int)(t / tilesN);
  int l = lane & 15, kb = (lane >> 4) << 3;
  int k = kt * 32 + kb + (h < 8 ? h : h + 8);
  int col = nt * 16 + l;
  Wp[i] = (k < Kreal) ? (_Float16)W[(long)k * C + col] : (_Float16)0.f;
}

// -------------------------------------------------------------------------
// Block-cooperative WMMA GEMM. Block = 8 waves = 128 rows x 64 cols; all
// waves share one 64-column weight panel staged in LDS (ktiles*4KB, <=64KB).
// Panel staged via Tensor Data Mover (tensor_load_to_lds + s_wait_tensorcnt,
// 6-arg clang-23 form); else cooperative b128 copy. Inner loop: 2 global
// b128 A loads + 8 ds_load_b128 B loads + 4 WMMAs, no branches -> EXEC stays
// all-ones. Output written as f16 (+bias); f32 per-channel sum/sumsq atomics
// feed the training-mode BN.
// -------------------------------------------------------------------------
__global__ void wmma_gemm_lds(const _Float16* __restrict__ A,
                              const _Float16* __restrict__ Wp,
                              const float* __restrict__ bias,
                              _Float16* __restrict__ Yo,
                              float* __restrict__ stats,
                              int M, int Kpad, int C) {
  extern __shared__ _Float16 Bs[];                 // ktiles * 2048 halves
  const int tilesN = C >> 4;
  const int ktiles = Kpad >> 5;
  const int nt0  = blockIdx.y << 2;
  const int wave = threadIdx.x >> 5;
  const int lane = threadIdx.x & 31;
  const int m0   = (blockIdx.x * 8 + wave) << 4;

#if __has_builtin(__builtin_amdgcn_tensor_load_to_lds)
  if (wave == 0) {
    // D# group0: count=1 | lds_addr(0: dynamic LDS base) | global_addr | type=2
    unsigned long long ga =
        (unsigned long long)(const void*)(Wp + (long)nt0 * 512);
    v4u g0;
    g0[0] = 1u;
    g0[1] = 0u;
    g0[2] = (unsigned)(ga & 0xffffffffu);
    g0[3] = (unsigned)((ga >> 32) & 0x01ffffffu) | (2u << 30);
    // D# group1: data_size=2B; 2D tensor: dim0 = tilesN*512 halves (stride
    // same), dim1 = ktiles; tile: 2048 x ktiles.
    unsigned td0 = (unsigned)tilesN << 9;
    unsigned td1 = (unsigned)ktiles;
    unsigned s0  = (unsigned)tilesN << 9;
    v8i g1;
    g1[0] = 0x10000;                                   // data_size=1 -> 2B
    g1[1] = (int)((td0 & 0xffffu) << 16);              // td0[15:0]
    g1[2] = (int)((td0 >> 16) | ((td1 & 0xffffu) << 16));
    g1[3] = (int)((td1 >> 16) | (2048u << 16));        // tile_dim0=2048
    g1[4] = (int)(td1 & 0xffffu);                      // tile_dim1=ktiles
    g1[5] = (int)s0;                                   // dim0 stride lo32
    g1[6] = 0;
    g1[7] = 0;
    v4i g2 = {0, 0, 0, 0}, g3 = {0, 0, 0, 0};
    v8i g4 = {0, 0, 0, 0, 0, 0, 0, 0};
    __builtin_amdgcn_tensor_load_to_lds(g0, g1, g2, g3, g4, 0);
    __builtin_amdgcn_s_wait_tensorcnt((short)0);
  }
  __syncthreads();
#else
  for (int idx = threadIdx.x; idx < (ktiles << 8); idx += 256) {
    const int kt = idx >> 8;
    const int off = (idx & 255) << 3;
    const _Float16* src = Wp + (long)(kt * tilesN + nt0) * 512 + off;
    *(v8h*)&Bs[(kt << 11) + off] = *(const v8h*)src;
  }
  __syncthreads();
#endif

  const int half = lane >> 4, l = lane & 15, kb = half << 3;
  const _Float16* __restrict__ arow = A + (long)(m0 + l) * Kpad;

  v8f z = {0.f, 0.f, 0.f, 0.f, 0.f, 0.f, 0.f, 0.f};
  v8f acc0 = z, acc1 = z, acc2 = z, acc3 = z;
  for (int kt = 0; kt < ktiles; ++kt) {
    const int k0 = kt << 5;
    v8h alo = *(const v8h*)(arow + k0 + kb);
    v8h ahi = *(const v8h*)(arow + k0 + kb + 16);
    v16h a = __builtin_shufflevector(alo, ahi,
        0, 1, 2, 3, 4, 5, 6, 7, 8, 9, 10, 11, 12, 13, 14, 15);
    const _Float16* bb = &Bs[(kt << 11) + lane * 16];
    v16h b0 = *(const v16h*)(bb);
    v16h b1 = *(const v16h*)(bb + 512);
    v16h b2 = *(const v16h*)(bb + 1024);
    v16h b3 = *(const v16h*)(bb + 1536);
    acc0 = __builtin_amdgcn_wmma_f32_16x16x32_f16(false, a, false, b0, (short)0, acc0, false, false);
    acc1 = __builtin_amdgcn_wmma_f32_16x16x32_f16(false, a, false, b1, (short)0, acc1, false, false);
    acc2 = __builtin_amdgcn_wmma_f32_16x16x32_f16(false, a, false, b2, (short)0, acc2, false, false);
    acc3 = __builtin_amdgcn_wmma_f32_16x16x32_f16(false, a, false, b3, (short)0, acc3, false, false);
  }

  v8f accs[4] = {acc0, acc1, acc2, acc3};
#pragma unroll
  for (int nt = 0; nt < 4; ++nt) {
    const int bcol = ((nt0 + nt) << 4) + l;
    const float bv = bias[bcol];
    float s = 0.f, sq = 0.f;
#pragma unroll
    for (int r = 0; r < 8; ++r) {
      float v = accs[nt][r] + bv;
      Yo[(long)(m0 + r + (half << 3)) * C + bcol] = (_Float16)v;
      s += v; sq += v * v;
    }
    atomicAdd(&stats[bcol], s);
    atomicAdd(&stats[C + bcol], sq);
  }
}

// In-place BN (training-mode: stats over all M rows) + ReLU on f16.
__global__ void bn_relu_f16(_Float16* __restrict__ Y, const float* __restrict__ stats,
                            const float* __restrict__ gamma, const float* __restrict__ beta,
                            long total, int C, float invM) {
  long i = (long)blockIdx.x * blockDim.x + threadIdx.x;
  if (i >= total) return;
  int ch = (int)(i % C);
  float mu  = stats[ch] * invM;
  float var = fmaxf(stats[C + ch] * invM - mu * mu, 0.f);
  float v = ((float)Y[i] - mu) * rsqrtf(var + 1e-5f) * gamma[ch] + beta[ch];
  Y[i] = (_Float16)fmaxf(v, 0.f);
}

__global__ void maxpool_f16(const _Float16* __restrict__ Y, float* __restrict__ out,
                            long rows, int K, int C) {
  long i = (long)blockIdx.x * blockDim.x + threadIdx.x;   // rows*C
  if (i >= rows * C) return;
  int c = (int)(i % C);
  long r = i / C;
  const _Float16* p = Y + (r * (long)K) * C + c;
  float m = (float)p[0];
  for (int k = 1; k < K; ++k) m = fmaxf(m, (float)p[(long)k * C]);
  out[i] = m;
}

// -------------------------------------------------------------------------
// Host orchestration. d_in layout (setup_inputs insertion order):
//   [0] xyz (16*8192*6); then per layer (sa1[0..2], sa2[0..2], sa3[0..2]):
//   W, b, gamma, beta.
// -------------------------------------------------------------------------
extern "C" void kernel_launch(void* const* d_in, const int* in_sizes, int n_in,
                              void* d_out, int out_size, void* d_ws, size_t ws_size,
                              hipStream_t stream) {
  (void)in_sizes; (void)n_in; (void)out_size; (void)ws_size;
  const float* xyz = (const float*)d_in[0];
  auto P = [&](int layer, int which) -> const float* {
    return (const float*)d_in[1 + layer * 4 + which];   // 0=W 1=b 2=gamma 3=beta
  };

  char* wsp = (char*)d_ws;
  auto alloc = [&](size_t bytes) -> void* {
    void* p = (void*)wsp;
    wsp += (bytes + 255) & ~(size_t)255;
    return p;
  };

  const int B = BATCH, N = NPTS;
  const int S1 = 512, K1 = 32, S2 = 128, S3 = 1, K2 = 64, K3 = 128;
  const long M1 = (long)B * S1 * K1;   // 262144
  const long M2 = (long)B * S2 * K2;   // 131072
  const long M3 = (long)B * S3 * K3;   // 2048

  int*      fidx1 = (int*)     alloc((size_t)B * S1 * 4);
  int*      gidx1 = (int*)     alloc((size_t)B * S1 * K1 * 4);
  int*      fidx2 = (int*)     alloc((size_t)B * S2 * 4);
  int*      gidx2 = (int*)     alloc((size_t)B * S2 * K2 * 4);
  float*    l1x   = (float*)   alloc((size_t)B * S1 * 3 * 4);
  float*    l2x   = (float*)   alloc((size_t)B * S2 * 3 * 4);
  float*    l1p   = (float*)   alloc((size_t)B * S1 * 128 * 4);
  float*    l2p   = (float*)   alloc((size_t)B * S2 * 256 * 4);
  float*    stats = (float*)   alloc((size_t)2 * 1024 * 4);
  _Float16* Wp    = (_Float16*)alloc((size_t)512 * 1024 * 2);   // max packed W
  _Float16* X16   = (_Float16*)alloc((size_t)M2 * 160 * 2);     // max grouped input
  _Float16* f16P  = (_Float16*)alloc((size_t)M1 * 128 * 2);     // ping
  _Float16* f16Q  = (_Float16*)alloc((size_t)M1 * 128 * 2);     // pong

  auto layer = [&](const _Float16* Ain, int Kpad, int Kreal,
                   _Float16* Aout, long M, int C, int li) {
    pack_w<<<(int)(((long)Kpad * C + 255) / 256), 256, 0, stream>>>(
        P(li, 0), Wp, Kreal, Kpad, C);
    zero_f32<<<(2 * C + 255) / 256, 256, 0, stream>>>(stats, 2 * C);
    dim3 g((unsigned)(M >> 7), (unsigned)(C >> 6));
    wmma_gemm_lds<<<g, 256, (size_t)Kpad * 128, stream>>>(
        Ain, Wp, P(li, 1), Aout, stats, (int)M, Kpad, C);
    long tot = M * (long)C;
    bn_relu_f16<<<(int)((tot + 255) / 256), 256, 0, stream>>>(
        Aout, stats, P(li, 2), P(li, 3), tot, C, 1.0f / (float)M);
  };

  // ---------------- SA1: npoint=512, r=0.2, ns=32, 6->64->64->128 ---------
  fps_kernel<<<B, FPS_T, 0, stream>>>(xyz, 6, N, S1, fidx1);
  gather_centroids<<<(B * S1 + 255) / 256, 256, 0, stream>>>(xyz, 6, N, fidx1, l1x, B * S1, S1);
  ball_query_kernel<<<(B * S1 * 32 + 255) / 256, 256, 0, stream>>>(
      xyz, 6, N, l1x, gidx1, B * S1, S1, K1, 0.04f);
  gather_group_f16<<<(int)((M1 + 255) / 256), 256, 0, stream>>>(
      xyz, 6, xyz + 3, 6, 3, l1x, gidx1, X16, N, S1, K1, 32, M1);
  layer(X16,  32,  6,  f16P, M1,  64, 0);
  layer(f16P, 64,  64, f16Q, M1,  64, 1);
  layer(f16Q, 64,  64, f16P, M1, 128, 2);
  maxpool_f16<<<(int)(((long)B * S1 * 128 + 255) / 256), 256, 0, stream>>>(
      f16P, l1p, (long)B * S1, K1, 128);

  // ---------------- SA2: npoint=128, r=0.4, ns=64, 131->128->128->256 -----
  fps_kernel<<<B, FPS_T, 0, stream>>>(l1x, 3, S1, S2, fidx2);
  gather_centroids<<<(B * S2 + 255) / 256, 256, 0, stream>>>(l1x, 3, S1, fidx2, l2x, B * S2, S2);
  ball_query_kernel<<<(B * S2 * 32 + 255) / 256, 256, 0, stream>>>(
      l1x, 3, S1, l2x, gidx2, B * S2, S2, K2, 0.16f);
  gather_group_f16<<<(int)((M2 + 255) / 256), 256, 0, stream>>>(
      l1x, 3, l1p, 128, 128, l2x, gidx2, X16, S1, S2, K2, 160, M2);
  layer(X16,  160, 131, f16P, M2, 128, 3);
  layer(f16P, 128, 128, f16Q, M2, 128, 4);
  layer(f16Q, 128, 128, f16P, M2, 256, 5);
  maxpool_f16<<<(int)(((long)B * S2 * 256 + 255) / 256), 256, 0, stream>>>(
      f16P, l2p, (long)B * S2, K2, 256);

  // ---------------- SA3: group_all, 259->256->512->1024 -------------------
  gather_group_f16<<<(int)((M3 + 255) / 256), 256, 0, stream>>>(
      l2x, 3, l2p, 256, 256, nullptr, nullptr, X16, S2, S3, K3, 288, M3);
  layer(X16,  288, 259, f16P, M3,  256, 6);
  layer(f16P, 256, 256, f16Q, M3,  512, 7);
  layer(f16Q, 512, 512, f16P, M3, 1024, 8);
  maxpool_f16<<<(int)(((long)B * 1024 + 255) / 256), 256, 0, stream>>>(
      f16P, (float*)d_out, (long)B, K3, 1024);
}
